// BiasedSelfAttnBlock_47768626266336
// MI455X (gfx1250) — compile-verified, matrix operand order checked by reading, not run on hardware
//
#include <hip/hip_runtime.h>

typedef __bf16 bf16_t;
typedef __attribute__((ext_vector_type(16))) __bf16 v16bf;
typedef __attribute__((ext_vector_type(8)))  float  v8f;

#define BATCH 4
#define CCH   512
#define HWSZ  4096
#define NGRP  32
#define CPG   16
#define GNEPS 1e-6f
#define QKSCALE 0.04419417382415922f   // 1/sqrt(512)

#define TQ 32
#define TK 64
#define SP 68   // S row pitch (floats)
#define PP 72   // P row pitch (bf16, multiple of 8 -> 16B-aligned rows)

__device__ __forceinline__ v8f zero8() {
    v8f v;
    for (int i = 0; i < 8; ++i) v[i] = 0.0f;
    return v;
}

// Load one 16x32 bf16 WMMA fragment from a row-major [rows, ld] array.
// Lane l (l<16): rows row+l, K runs {0..7, 16..23}; lanes 16..31: K runs {8..15, 24..31}.
// Each run is 8 contiguous bf16 = one b128 load.
__device__ __forceinline__ v16bf load_frag(const bf16_t* __restrict__ base, int row, int ld, int k0) {
    const int lane = threadIdx.x & 31;
    const int l16  = lane & 15;
    const int half = lane >> 4;
    const bf16_t* p = base + (size_t)(row + l16) * ld + k0 + half * 8;
    union { uint4 q[2]; v16bf v; } u;
    u.q[0] = *reinterpret_cast<const uint4*>(p);
    u.q[1] = *reinterpret_cast<const uint4*>(p + 16);
    return u.v;
}

__device__ __forceinline__ v8f wmma_bf16(v16bf a, v16bf b, v8f c) {
    return __builtin_amdgcn_wmma_f32_16x16x32_bf16(false, a, false, b, (short)0, c, false, false);
}

// ---------------------------------------------------------------------------
// fp32 -> bf16 convert (for weight matrices)
// ---------------------------------------------------------------------------
__global__ __launch_bounds__(256)
void f32_to_bf16_kernel(const float* __restrict__ src, bf16_t* __restrict__ dst, int n) {
    int i = blockIdx.x * blockDim.x + threadIdx.x;
    if (i < n) dst[i] = (bf16_t)src[i];
}

// ---------------------------------------------------------------------------
// GroupNorm: x [B,C,H,W] fp32 -> hn^T [B, HW, C] bf16
// One block per (group, batch). Each group's slab (16ch x 4096) is contiguous.
// ---------------------------------------------------------------------------
__global__ __launch_bounds__(256)
void groupnorm_kernel(const float* __restrict__ x, const float* __restrict__ gamma,
                      const float* __restrict__ beta, bf16_t* __restrict__ hnT) {
    __shared__ float rs[256];
    __shared__ float rq[256];
    const int g = blockIdx.x, b = blockIdx.y, tid = threadIdx.x;
    const int c0 = g * CPG;
    const float* xb = x + ((size_t)b * CCH + c0) * HWSZ;
    const int NE = CPG * HWSZ;

    float s = 0.f, q = 0.f;
    for (int i = tid; i < NE; i += 256) { float v = xb[i]; s += v; q += v * v; }
    rs[tid] = s; rq[tid] = q;
    __syncthreads();
    for (int st = 128; st > 0; st >>= 1) {
        if (tid < st) { rs[tid] += rs[tid + st]; rq[tid] += rq[tid + st]; }
        __syncthreads();
    }
    const float inv_n = 1.0f / (float)NE;
    const float mean  = rs[0] * inv_n;
    const float var   = rq[0] * inv_n - mean * mean;
    const float rstd  = rsqrtf(var + GNEPS);

    // write transposed: 16-thread groups write 32B-contiguous chunks of hn^T
    for (int i = tid; i < NE; i += 256) {
        const int cl = i & (CPG - 1);
        const int n  = i >> 4;
        const float v = xb[(size_t)cl * HWSZ + n];
        const float h = (v - mean) * rstd * gamma[c0 + cl] + beta[c0 + cl];
        hnT[((size_t)b * HWSZ + n) * CCH + c0 + cl] = (bf16_t)h;
    }
}

// ---------------------------------------------------------------------------
// WMMA GEMM: out[o,n] = sum_c A[o,c] * Bt[n,c] + bias[o]
// A: [512,512] bf16 weights; Bt: [B, HW, 512] bf16 (K-contiguous).
// Block tile 64(M) x 256(N); 8 waves, each 32x64 = 2x4 accumulators.
// OUT_TRANSPOSED=1 -> out [B, HW, 512] (packed 16B stores); else [B, 512, HW].
// ---------------------------------------------------------------------------
template <int OUT_TRANSPOSED>
__global__ __launch_bounds__(256)
void gemm_qkv_kernel(const bf16_t* __restrict__ A, const bf16_t* __restrict__ Bt,
                     const float* __restrict__ bias, bf16_t* __restrict__ out) {
    const int b    = blockIdx.z;
    const int wave = threadIdx.x >> 5;
    const int lane = threadIdx.x & 31;
    const int l16  = lane & 15, half = lane >> 4;
    const int m0w  = blockIdx.x * 64  + (wave >> 2) * 32;
    const int n0w  = blockIdx.y * 256 + (wave & 3) * 64;
    const bf16_t* Btb = Bt + (size_t)b * HWSZ * CCH;

    v8f acc[2][4];
    for (int i = 0; i < 2; ++i)
        for (int j = 0; j < 4; ++j) acc[i][j] = zero8();

    for (int k0 = 0; k0 < CCH; k0 += 32) {
        v16bf a0 = load_frag(A, m0w,      CCH, k0);
        v16bf a1 = load_frag(A, m0w + 16, CCH, k0);
        for (int j = 0; j < 4; ++j) {
            v16bf bb = load_frag(Btb, n0w + j * 16, CCH, k0);
            acc[0][j] = wmma_bf16(a0, bb, acc[0][j]);
            acc[1][j] = wmma_bf16(a1, bb, acc[1][j]);
        }
    }

    if (OUT_TRANSPOSED) {
        for (int i = 0; i < 2; ++i)
            for (int j = 0; j < 4; ++j) {
                const int obase = m0w + i * 16 + half * 8;
                const int n     = n0w + j * 16 + l16;
                union { uint4 q; bf16_t e[8]; } u;
                for (int r = 0; r < 8; ++r) u.e[r] = (bf16_t)(acc[i][j][r] + bias[obase + r]);
                *reinterpret_cast<uint4*>(out + ((size_t)b * HWSZ + n) * CCH + obase) = u.q;
            }
    } else {
        for (int i = 0; i < 2; ++i)
            for (int j = 0; j < 4; ++j) {
                const int n = n0w + j * 16 + l16;
                for (int r = 0; r < 8; ++r) {
                    const int o = m0w + i * 16 + half * 8 + r;
                    out[((size_t)b * CCH + o) * HWSZ + n] = (bf16_t)(acc[i][j][r] + bias[o]);
                }
            }
    }
}

// ---------------------------------------------------------------------------
// Flash attention with post-softmax mask folded in (denominator cancels):
//   attn = mask*exp(s) / sum(mask*exp(s))
// Per block: 32 queries, loop over 64-key tiles. Single pass, no score tensor.
// qT,kT: [B,HW,C]; v: [B,C,HW]; mask: [B,HW]; oT out: [B,HW,C].
// ---------------------------------------------------------------------------
__global__ __launch_bounds__(256)
void flash_attn_kernel(const bf16_t* __restrict__ qT, const bf16_t* __restrict__ kT,
                       const bf16_t* __restrict__ v,  const float* __restrict__ mask,
                       bf16_t* __restrict__ oT) {
    __shared__ float  S[TQ][SP];
    __shared__ __align__(16) bf16_t P[TQ][PP];
    __shared__ float  mTile[TK];
    __shared__ float  corr[TQ];

    const int b    = blockIdx.y;
    const int q0   = blockIdx.x * TQ;
    const int tid  = threadIdx.x;
    const int wave = tid >> 5, lane = tid & 31;
    const int l16  = lane & 15, half = lane >> 4;

    const bf16_t* qb = qT + (size_t)b * HWSZ * CCH + (size_t)q0 * CCH;
    const bf16_t* kb = kT + (size_t)b * HWSZ * CCH;
    const bf16_t* vb = v  + (size_t)b * CCH * HWSZ;

    const int smi = wave >> 2, sni = wave & 3; // S-tile ownership (32x64 -> 2x4 tiles)
    const int cw  = wave * 64;                 // O column range per wave (512/8)

    v8f oacc[2][4];
    for (int i = 0; i < 2; ++i)
        for (int j = 0; j < 4; ++j) oacc[i][j] = zero8();

    float m_run = -3.0e38f, l_run = 0.0f;

    for (int k0 = 0; k0 < HWSZ; k0 += TK) {
        // ---- S tile: per-wave 16x16, K = 512 ----
        v8f sacc = zero8();
        for (int kk = 0; kk < CCH; kk += 32) {
            v16bf a  = load_frag(qb, smi * 16, CCH, kk);
            v16bf bb = load_frag(kb + (size_t)k0 * CCH, sni * 16, CCH, kk);
            sacc = wmma_bf16(a, bb, sacc);
        }
        __syncthreads();  // previous iteration's P/S consumption complete
        for (int r = 0; r < 8; ++r)
            S[smi * 16 + half * 8 + r][sni * 16 + l16] = sacc[r] * QKSCALE;
        if (tid < TK) mTile[tid] = mask[(size_t)b * HWSZ + k0 + tid];
        __syncthreads();

        // ---- online masked softmax: wave 0, one lane per query row ----
        if (wave == 0) {
            float tm = -3.0e38f;
            for (int j = 0; j < TK; ++j) tm = fmaxf(tm, S[lane][j]);
            const float m_new = fmaxf(m_run, tm);
            const float cr    = __expf(m_run - m_new);
            float ps = 0.0f;
            for (int j = 0; j < TK; ++j) {
                const float p = mTile[j] * __expf(S[lane][j] - m_new);
                P[lane][j] = (bf16_t)p;
                ps += p;
            }
            l_run = l_run * cr + ps;
            m_run = m_new;
            corr[lane] = cr;
        }
        __syncthreads();

        // ---- rescale O accumulators, then O += P * v^T (K = 64 keys) ----
        for (int i = 0; i < 2; ++i)
            for (int r = 0; r < 8; ++r) {
                const float cr = corr[i * 16 + half * 8 + r];
                for (int j = 0; j < 4; ++j) oacc[i][j][r] *= cr;
            }
        for (int ks = 0; ks < TK; ks += 32) {
            v16bf a0 = load_frag(&P[0][0], 0,  PP, ks);
            v16bf a1 = load_frag(&P[0][0], 16, PP, ks);
            for (int j = 0; j < 4; ++j) {
                v16bf bv = load_frag(vb, cw + j * 16, HWSZ, k0 + ks);
                oacc[0][j] = wmma_bf16(a0, bv, oacc[0][j]);
                oacc[1][j] = wmma_bf16(a1, bv, oacc[1][j]);
            }
        }
    }

    // ---- finalize: divide by running sum, store O^T [HW, C] bf16 ----
    if (wave == 0) corr[lane] = 1.0f / l_run;
    __syncthreads();
    bf16_t* ob = oT + (size_t)b * HWSZ * CCH;
    for (int i = 0; i < 2; ++i)
        for (int r = 0; r < 8; ++r) {
            const int qm = i * 16 + half * 8 + r;
            const float inv = corr[qm];
            for (int j = 0; j < 4; ++j) {
                const int c = cw + j * 16 + l16;
                ob[(size_t)(q0 + qm) * CCH + c] = (bf16_t)(oacc[i][j][r] * inv);
            }
        }
}

// ---------------------------------------------------------------------------
// Output projection + residual: out = x + Wo * O + bo  (fp32 epilogue)
// ---------------------------------------------------------------------------
__global__ __launch_bounds__(256)
void gemm_out_kernel(const bf16_t* __restrict__ Wo, const bf16_t* __restrict__ Ot,
                     const float* __restrict__ bo, const float* __restrict__ x,
                     float* __restrict__ out) {
    const int b    = blockIdx.z;
    const int wave = threadIdx.x >> 5;
    const int lane = threadIdx.x & 31;
    const int l16  = lane & 15, half = lane >> 4;
    const int m0w  = blockIdx.x * 64  + (wave >> 2) * 32;
    const int n0w  = blockIdx.y * 256 + (wave & 3) * 64;
    const bf16_t* Ob = Ot + (size_t)b * HWSZ * CCH;

    v8f acc[2][4];
    for (int i = 0; i < 2; ++i)
        for (int j = 0; j < 4; ++j) acc[i][j] = zero8();

    for (int k0 = 0; k0 < CCH; k0 += 32) {
        v16bf a0 = load_frag(Wo, m0w,      CCH, k0);
        v16bf a1 = load_frag(Wo, m0w + 16, CCH, k0);
        for (int j = 0; j < 4; ++j) {
            v16bf bb = load_frag(Ob, n0w + j * 16, CCH, k0);
            acc[0][j] = wmma_bf16(a0, bb, acc[0][j]);
            acc[1][j] = wmma_bf16(a1, bb, acc[1][j]);
        }
    }

    for (int i = 0; i < 2; ++i)
        for (int j = 0; j < 4; ++j) {
            const int n = n0w + j * 16 + l16;
            for (int r = 0; r < 8; ++r) {
                const int o = m0w + i * 16 + half * 8 + r;
                const size_t a = ((size_t)b * CCH + o) * HWSZ + n;
                out[a] = x[a] + bo[o] + acc[i][j][r];
            }
        }
}

// ---------------------------------------------------------------------------
extern "C" void kernel_launch(void* const* d_in, const int* in_sizes, int n_in,
                              void* d_out, int out_size, void* d_ws, size_t ws_size,
                              hipStream_t stream) {
    const float* x     = (const float*)d_in[0];
    const float* mask  = (const float*)d_in[1];
    const float* gamma = (const float*)d_in[2];
    const float* beta  = (const float*)d_in[3];
    const float* Wq = (const float*)d_in[4];
    const float* bq = (const float*)d_in[5];
    const float* Wk = (const float*)d_in[6];
    const float* bk = (const float*)d_in[7];
    const float* Wv = (const float*)d_in[8];
    const float* bv = (const float*)d_in[9];
    const float* Wo = (const float*)d_in[10];
    const float* bo = (const float*)d_in[11];
    float* out = (float*)d_out;

    // workspace layout (bf16): needs ~69 MB
    bf16_t* w = (bf16_t*)d_ws;
    const size_t MAT = (size_t)BATCH * HWSZ * CCH;     // 8.4M elems
    bf16_t* hnT = w;                                   // [B,HW,C]; reused as attn-out^T
    bf16_t* qT  = w + MAT;
    bf16_t* kT  = w + 2 * MAT;
    bf16_t* vW  = w + 3 * MAT;                         // [B,C,HW]
    bf16_t* wq  = w + 4 * MAT;
    bf16_t* wk  = wq + (size_t)CCH * CCH;
    bf16_t* wv  = wk + (size_t)CCH * CCH;
    bf16_t* wo  = wv + (size_t)CCH * CCH;

    const int WN = CCH * CCH;
    f32_to_bf16_kernel<<<WN / 256, 256, 0, stream>>>(Wq, wq, WN);
    f32_to_bf16_kernel<<<WN / 256, 256, 0, stream>>>(Wk, wk, WN);
    f32_to_bf16_kernel<<<WN / 256, 256, 0, stream>>>(Wv, wv, WN);
    f32_to_bf16_kernel<<<WN / 256, 256, 0, stream>>>(Wo, wo, WN);

    groupnorm_kernel<<<dim3(NGRP, BATCH), 256, 0, stream>>>(x, gamma, beta, hnT);

    dim3 gg(CCH / 64, HWSZ / 256, BATCH);
    gemm_qkv_kernel<1><<<gg, 256, 0, stream>>>(wq, hnT, bq, qT);
    gemm_qkv_kernel<1><<<gg, 256, 0, stream>>>(wk, hnT, bk, kT);
    gemm_qkv_kernel<0><<<gg, 256, 0, stream>>>(wv, hnT, bv, vW);

    flash_attn_kernel<<<dim3(HWSZ / TQ, BATCH), 256, 0, stream>>>(qT, kT, vW, mask, hnT);

    gemm_out_kernel<<<gg, 256, 0, stream>>>(wo, hnT, bo, x, out);
}
